// myGRU_29987461660952
// MI455X (gfx1250) — matrix operand order repaired and checked by reading, not compiled, not run on hardware
//
#include <hip/hip_runtime.h>

#define DEV __device__ __forceinline__

typedef __attribute__((ext_vector_type(16))) __bf16 v16bf;
typedef __attribute__((ext_vector_type(8)))  __bf16 v8bf;
typedef __attribute__((ext_vector_type(8)))  float  v8f;
typedef unsigned int u32;
typedef __attribute__((ext_vector_type(4))) u32 u32x4;
typedef __attribute__((ext_vector_type(8))) int  i32x8;
typedef __attribute__((ext_vector_type(4))) int  i32x4;

#if defined(__has_builtin)
#  if __has_builtin(__builtin_amdgcn_tensor_load_to_lds) && __has_builtin(__builtin_amdgcn_s_wait_tensorcnt)
#    define USE_TDM 1
#  else
#    define USE_TDM 0
#  endif
#else
#  define USE_TDM 0
#endif

namespace {
constexpr int B = 32, S = 512, D = 1024, H = 1024;
constexpr int G  = 3 * H;             // 3072 gate columns
constexpr long MROWS = (long)B * S;   // 16384 rows of the input-projection GEMM
constexpr int NT_G = G / 16;          // 192 n-tiles over the 3H gate dim
constexpr int KC_D = D / 32;          // 32 K-chunks (input projection)
constexpr int KC_H = H / 32;          // 32 K-chunks (hidden projection)
constexpr int NB   = 64;              // persistent blocks (one 16-col slab of H each)
}

// ---------- scalar bf16 helpers (round-to-nearest-even) ----------
DEV unsigned short f2bf(float f) {
  unsigned u = __float_as_uint(f);
  u += 0x7FFFu + ((u >> 16) & 1u);
  return (unsigned short)(u >> 16);
}
DEV float bf2f(unsigned short s) { return __uint_as_float(((unsigned)s) << 16); }

// ---------- WMMA A-fragment load (16x32 bf16, row-major source) ----------
DEV v16bf load_a(const unsigned short* __restrict__ row, int kbase, int half) {
  union { v16bf v; struct { v8bf lo, hi; } s; } a;
  a.s.lo = *(const v8bf*)(row + kbase + half * 8);
  a.s.hi = *(const v8bf*)(row + kbase + 16 + half * 8);
  return a.v;
}

// ---------- device-wide split barrier (monotonic counter) ----------
DEV void grid_sync(unsigned* cnt, unsigned target) {
  __threadfence();                       // release this thread's stores to device scope
  __syncthreads();
  if (threadIdx.x == 0) {
    __hip_atomic_fetch_add(cnt, 1u, __ATOMIC_RELEASE, __HIP_MEMORY_SCOPE_AGENT);
    while (__hip_atomic_load(cnt, __ATOMIC_ACQUIRE, __HIP_MEMORY_SCOPE_AGENT) < target)
      __builtin_amdgcn_s_sleep(2);
  }
  __syncthreads();
  __threadfence();                       // acquire side
}

#if USE_TDM
// ---------- TDM: load a 2D tile (tile_dim0=16 cols x rows) of bf16 into LDS ----------
// row stride in elements; gaddr = byte address of tile start.
DEV void tdm_load_tile(unsigned lds_off, unsigned long long gaddr,
                       int rows, unsigned row_stride_elems) {
  u32x4 g0;
  g0[0] = 1u;                                             // count=1, user descriptor
  g0[1] = lds_off;                                        // lds_addr (bytes)
  g0[2] = (u32)gaddr;                                     // global_addr[31:0]
  g0[3] = ((u32)(gaddr >> 32) & 0x01FFFFFFu) | (2u << 30); // addr[56:32] | type=2
  const u32 TD0 = 0x7FFFFFFFu;                            // tensor_dim0 (huge: no OOB)
  const u32 TD1 = (u32)rows;                              // tensor_dim1
  i32x8 g1;
  g1[0] = (int)(1u << 16);                                // wg_mask=0, data_size=1 (2B)
  g1[1] = (int)((TD0 & 0xFFFFu) << 16);                   // atomic_barrier=0 | TD0[15:0]
  g1[2] = (int)((TD0 >> 16) | ((TD1 & 0xFFFFu) << 16));   // TD0[31:16] | TD1[15:0]
  g1[3] = (int)((TD1 >> 16) | (16u << 16));               // TD1[31:16] | tile_dim0=16
  g1[4] = (int)((u32)rows);                               // tile_dim1=rows, tile_dim2=0
  g1[5] = (int)row_stride_elems;                          // tensor_dim0_stride[31:0]
  g1[6] = 0;                                              // stride hi | dim1_stride lo
  g1[7] = 0;
  i32x4 gz4 = {0, 0, 0, 0};
  i32x8 gz8 = {0, 0, 0, 0, 0, 0, 0, 0};
  // 6-arg toolchain form: (g0, g1, g2, g3, g4, cpol)
  __builtin_amdgcn_tensor_load_to_lds(g0, g1, gz4, gz4, gz8, 0);
}
#endif

// ---------- fp32 -> bf16 bulk convert ----------
__global__ void k_cvt_bf16(const float* __restrict__ x, unsigned short* __restrict__ y, long n) {
  long i = (long)blockIdx.x * blockDim.x + threadIdx.x;
  long stride = (long)gridDim.x * blockDim.x;
  for (; i < n; i += stride) y[i] = f2bf(x[i]);
}

// ---------- pack weight [nrows x K] row-major into WMMA B-fragment layout ----------
__global__ void k_pack_w(const float* __restrict__ W, unsigned short* __restrict__ P,
                         int nrows, int K) {
  const int KC = K / 32;
  long total = (long)nrows * K;
  long i = (long)blockIdx.x * blockDim.x + threadIdx.x;
  long stride = (long)gridDim.x * blockDim.x;
  for (; i < total; i += stride) {
    int e    = (int)(i & 15);
    int lane = (int)((i >> 4) & 31);
    long r   = i >> 9;
    int kc   = (int)(r % KC);
    int nt   = (int)(r / KC);
    P[i] = f2bf(W[(long)(nt * 16 + (lane & 15)) * K + kc * 32 + ((e & 8) << 1) + (lane >> 4) * 8 + (e & 7)]);
  }
}

// ---------- zero initial state + barrier counter ----------
__global__ void k_init(float* __restrict__ hf, unsigned short* __restrict__ hb,
                       unsigned* __restrict__ bar) {
  int i = blockIdx.x * blockDim.x + threadIdx.x;
  if (i < B * H) { hf[i] = 0.0f; hb[i] = 0; }
  if (i == 0) *bar = 0u;
}

// ---------- phase 1: gi = feats @ W_ih^T + b_ih, stored bf16 [MROWS, 3H] ----------
__global__ __launch_bounds__(128) void k_gi_gemm(const unsigned short* __restrict__ A,
                                                 const unsigned short* __restrict__ Bp,
                                                 const float* __restrict__ bias,
                                                 unsigned short* __restrict__ gi) {
  const int lane = threadIdx.x & 31;
  const int w    = threadIdx.x >> 5;
  const int nt   = blockIdx.x;             // 0..191
  const int mt   = blockIdx.y * 4 + w;     // 0..1023
  const int half = lane >> 4;
  const unsigned short* Arow = A + (long)(mt * 16 + (lane & 15)) * D;
  const unsigned short* Bt   = Bp + (long)nt * KC_D * 512 + lane * 16;
  v8f acc = {};
  for (int kc = 0; kc < KC_D; ++kc) {
    if (kc + 1 < KC_D) __builtin_prefetch(Bt + (long)(kc + 1) * 512, 0, 0);
    v16bf a = load_a(Arow, kc * 32, half);
    v16bf b = *(const v16bf*)(Bt + (long)kc * 512);
    acc = __builtin_amdgcn_wmma_f32_16x16x32_bf16(false, a, false, b,
                                                  (short)0, acc, false, false);
  }
  const int n  = nt * 16 + (lane & 15);
  const float bv = bias[n];
#pragma unroll
  for (int v = 0; v < 8; ++v) {
    int m = mt * 16 + v + 8 * half;
    __builtin_nontemporal_store(f2bf(acc[v] + bv), &gi[(long)m * G + n]);
  }
}

// ---------- phase 2: persistent cooperative GRU scan ----------
// 64 blocks x 256 threads. Block nt owns H columns [nt*16, nt*16+16).
// Wave w holds W_hh K-chunks [w*4, w*4+4) for all 3 gates in registers (96 VGPRs)
// for the whole sequence. Per step: 24 WMMAs/wave, LDS reduction, 8-way split
// gate math, device-wide barrier. gi tiles for step t+1 prefetched via TDM.
__global__ __launch_bounds__(256) void k_gru_scan(
    unsigned short* __restrict__ hb0, unsigned short* __restrict__ hb1,
    float* __restrict__ hf0, float* __restrict__ hf1,
    const unsigned short* __restrict__ Bp,   // packed W_hh
    const float* __restrict__ bias,          // bias_hh [3H]
    const unsigned short* __restrict__ gi,   // bf16 [MROWS, 3H]
    float* __restrict__ out,                 // [B,S,H]
    unsigned* __restrict__ bar) {
  __shared__ float red[8][3][2][8][32];      // [wave][gate][mt][v][lane]  48KB
#if USE_TDM
  __shared__ unsigned short gis[2][3][B][16]; // staged gi tiles, 6KB, dbl-buffered
#endif
  const int lane = threadIdx.x & 31;
  const int w    = threadIdx.x >> 5;         // 0..7
  const int nt   = blockIdx.x;               // 0..63
  const int half = lane >> 4;

  // ---- register-resident W_hh fragments for this wave's K range ----
  v16bf bw[3][4];
#pragma unroll
  for (int g = 0; g < 3; ++g) {
    const unsigned short* base = Bp + (long)(nt + 64 * g) * KC_H * 512 + lane * 16;
#pragma unroll
    for (int c = 0; c < 4; ++c)
      bw[g][c] = *(const v16bf*)(base + (long)(w * 4 + c) * 512);
  }

  const int j = nt * 16 + (lane & 15);
  const float bs0 = bias[j], bs1 = bias[j + H], bs2 = bias[j + 2 * H];

#if USE_TDM
  if (w == 0) {
    // prefetch gi tiles for t=0 into buffer 0 (rows = all 32 batches, stride S*G)
#pragma unroll
    for (int g = 0; g < 3; ++g) {
      unsigned long long ga = (unsigned long long)(const void*)
          (gi + (long)0 * G + nt * 16 + g * H);           // gi row for (b=0, t=0)
      unsigned lds_off = (u32)(unsigned long long)(const void*)&gis[0][g][0][0];
      tdm_load_tile(lds_off, ga, B, (u32)((long)S * G));
    }
  }
#endif

  for (int t = 0; t < S; ++t) {
    const unsigned short* hbc = (t & 1) ? hb1 : hb0;
    const float*          hfc = (t & 1) ? hf1 : hf0;
    unsigned short*       hbn = (t & 1) ? hb0 : hb1;
    float*                hfn = (t & 1) ? hf0 : hf1;

#if USE_TDM
    if (w == 0 && t + 1 < S) {                  // prefetch next step's gi tiles
#pragma unroll
      for (int g = 0; g < 3; ++g) {
        unsigned long long ga = (unsigned long long)(const void*)
            (gi + (long)(t + 1) * G + nt * 16 + g * H);
        unsigned lds_off = (u32)(unsigned long long)(const void*)&gis[(t + 1) & 1][g][0][0];
        tdm_load_tile(lds_off, ga, B, (u32)((long)S * G));
      }
    }
#endif

    // ---- recurrent GEMM: gh tiles for both M tiles, 3 gates ----
    const unsigned short* Ar0 = hbc + (long)(lane & 15) * H;
    const unsigned short* Ar1 = hbc + (long)(16 + (lane & 15)) * H;
    v8f a00 = {}, a01 = {}, a02 = {}, a10 = {}, a11 = {}, a12 = {};
#pragma unroll
    for (int c = 0; c < 4; ++c) {
      const int kc = w * 4 + c;
      v16bf x0 = load_a(Ar0, kc * 32, half);
      v16bf x1 = load_a(Ar1, kc * 32, half);
      a00 = __builtin_amdgcn_wmma_f32_16x16x32_bf16(false, x0, false, bw[0][c], (short)0, a00, false, false);
      a10 = __builtin_amdgcn_wmma_f32_16x16x32_bf16(false, x1, false, bw[0][c], (short)0, a10, false, false);
      a01 = __builtin_amdgcn_wmma_f32_16x16x32_bf16(false, x0, false, bw[1][c], (short)0, a01, false, false);
      a11 = __builtin_amdgcn_wmma_f32_16x16x32_bf16(false, x1, false, bw[1][c], (short)0, a11, false, false);
      a02 = __builtin_amdgcn_wmma_f32_16x16x32_bf16(false, x0, false, bw[2][c], (short)0, a02, false, false);
      a12 = __builtin_amdgcn_wmma_f32_16x16x32_bf16(false, x1, false, bw[2][c], (short)0, a12, false, false);
    }
#pragma unroll
    for (int v = 0; v < 8; ++v) {
      red[w][0][0][v][lane] = a00[v];  red[w][1][0][v][lane] = a01[v];  red[w][2][0][v][lane] = a02[v];
      red[w][0][1][v][lane] = a10[v];  red[w][1][1][v][lane] = a11[v];  red[w][2][1][v][lane] = a12[v];
    }
#if USE_TDM
    if (w == 0) {
      if (t + 1 < S) __builtin_amdgcn_s_wait_tensorcnt(3);   // step-t tiles landed
      else           __builtin_amdgcn_s_wait_tensorcnt(0);
    }
#endif
    __syncthreads();

    // ---- reduce partials for v == w, then fused gate math (8-way split) ----
    float s00 = 0.f, s01 = 0.f, s02 = 0.f, s10 = 0.f, s11 = 0.f, s12 = 0.f;
#pragma unroll
    for (int ww = 0; ww < 8; ++ww) {
      s00 += red[ww][0][0][w][lane];  s01 += red[ww][1][0][w][lane];  s02 += red[ww][2][0][w][lane];
      s10 += red[ww][0][1][w][lane];  s11 += red[ww][1][1][w][lane];  s12 += red[ww][2][1][w][lane];
    }
#pragma unroll
    for (int mt = 0; mt < 2; ++mt) {
      const int b  = mt * 16 + w + 8 * half;       // C-layout row for v=w
      float h_r = (mt ? s10 : s00) + bs0;
      float h_i = (mt ? s11 : s01) + bs1;
      float h_n = (mt ? s12 : s02) + bs2;
      float i_r, i_i, i_n;
#if USE_TDM
      i_r = bf2f(gis[t & 1][0][b][lane & 15]);
      i_i = bf2f(gis[t & 1][1][b][lane & 15]);
      i_n = bf2f(gis[t & 1][2][b][lane & 15]);
#else
      const long gr = ((long)b * S + t) * G;
      i_r = bf2f(gi[gr + j]);
      i_i = bf2f(gi[gr + j + H]);
      i_n = bf2f(gi[gr + j + 2 * H]);
#endif
      const float rg = 1.0f / (1.0f + __expf(-(i_r + h_r)));
      const float ig = 1.0f / (1.0f + __expf(-(i_i + h_i)));
      const float ng = tanhf(i_n + rg * h_n);
      const float hp = hfc[b * H + j];
      const float hy = ng + ig * (hp - ng);
      __builtin_nontemporal_store(hy, &out[((long)b * S + t) * H + j]);
      hfn[b * H + j] = hy;
      hbn[b * H + j] = f2bf(hy);
    }

    grid_sync(bar, (unsigned)NB * (unsigned)(t + 1));
  }
}

extern "C" void kernel_launch(void* const* d_in, const int* in_sizes, int n_in,
                              void* d_out, int out_size, void* d_ws, size_t ws_size,
                              hipStream_t stream) {
  (void)in_sizes; (void)n_in; (void)out_size; (void)ws_size;
  const float* feats = (const float*)d_in[0];
  const float* w_ih  = (const float*)d_in[1];
  const float* w_hh  = (const float*)d_in[2];
  const float* b_ih  = (const float*)d_in[3];
  const float* b_hh  = (const float*)d_in[4];
  float* out = (float*)d_out;

  // ---- workspace carve (~141 MB) ----
  char* p = (char*)d_ws;
  auto take = [&](size_t bytes) { void* q = (void*)p; p += ((bytes + 255) & ~(size_t)255); return q; };
  unsigned short* featsb = (unsigned short*)take(MROWS * (size_t)D * 2);
  unsigned short* Pih    = (unsigned short*)take((size_t)G * D * 2);
  unsigned short* Phh    = (unsigned short*)take((size_t)G * H * 2);
  unsigned short* gi     = (unsigned short*)take(MROWS * (size_t)G * 2);
  float*          hf0    = (float*)take((size_t)B * H * 4);
  float*          hf1    = (float*)take((size_t)B * H * 4);
  unsigned short* hb0    = (unsigned short*)take((size_t)B * H * 2);
  unsigned short* hb1    = (unsigned short*)take((size_t)B * H * 2);
  unsigned*       bar    = (unsigned*)take(256);

  k_cvt_bf16<<<8192, 256, 0, stream>>>(feats, featsb, MROWS * (long)D);
  k_pack_w<<<6144, 256, 0, stream>>>(w_ih, Pih, G, D);
  k_pack_w<<<6144, 256, 0, stream>>>(w_hh, Phh, G, H);
  k_init<<<128, 256, 0, stream>>>(hf0, hb0, bar);

  k_gi_gemm<<<dim3(NT_G, MROWS / 16 / 4), 128, 0, stream>>>(featsb, Pih, b_ih, gi);

  k_gru_scan<<<NB, 256, 0, stream>>>(hb0, hb1, hf0, hf1, Phh, b_hh, gi, out, bar);

  (void)hipMemcpyAsync(out + MROWS * (long)H, hf0, (size_t)B * H * sizeof(float),
                       hipMemcpyDeviceToDevice, stream);
}
